// JSC_S_6528350290491
// MI455X (gfx1250) — 8-bit matrix operand order repaired and checked by reading, compiled, not run on hardware
//
#include <hip/hip_runtime.h>

typedef int v8i __attribute__((ext_vector_type(8)));

#define BN_EPS 1e-5f

__device__ __forceinline__ int q8(float x) {
    float r = rintf(x);
    r = fminf(fmaxf(r, -128.0f), 127.0f);
    return (int)r;
}

__device__ __forceinline__ unsigned pk4(int b0, int b1, int b2, int b3) {
    return (unsigned)(b0 & 255) | ((unsigned)(b1 & 255) << 8) |
           ((unsigned)(b2 & 255) << 16) | ((unsigned)(b3 & 255) << 24);
}

__device__ __forceinline__ void lds_fence() {
    asm volatile("s_wait_dscnt 0" ::: "memory");
}

__device__ __forceinline__ v8i wmma_iu8(v8i a, v8i b, v8i c) {
    return __builtin_amdgcn_wmma_i32_16x16x64_iu8(true, a, true, b, c, false, false);
}

__device__ v8i build_b(const float* __restrict__ w, int Nw, int Kw,
                       int n_base, int l15, int hi) {
    v8i o = {0, 0, 0, 0, 0, 0, 0, 0};
    const int n = n_base + l15;
    #pragma unroll
    for (int v = 0; v < 8; ++v) {
        int k0 = 16 * (v >> 1) + 8 * hi + 4 * (v & 1);
        int b[4];
        #pragma unroll
        for (int j = 0; j < 4; ++j) {
            int k = k0 + j;
            b[j] = (n < Nw && k < Kw) ? q8(w[n * Kw + k]) : 0;
        }
        o[v] = (int)pk4(b[0], b[1], b[2], b[3]);
    }
    return o;
}

__device__ __forceinline__ v8i layer_int(v8i a, const v8i* Bt, char* ldsb,
                                         int l15, int hi) {
    const v8i zero = {0, 0, 0, 0, 0, 0, 0, 0};
    #pragma unroll
    for (int t = 0; t < 2; ++t) {
        v8i c = wmma_iu8(a, Bt[t], zero);
        #pragma unroll
        for (int r = 0; r < 8; ++r) {
            int q = c[r];
            q = q < 0 ? 0 : (q > 127 ? 127 : q);
            ldsb[(r + 8 * hi) * 32 + 16 * t + l15] = (char)q;
        }
    }
    lds_fence();
    v8i an = {0, 0, 0, 0, 0, 0, 0, 0};
    #pragma unroll
    for (int v = 0; v < 4; ++v) {
        int k0 = 16 * (v >> 1) + 8 * hi + 4 * (v & 1);
        an[v] = *(const int*)(ldsb + l15 * 32 + k0);
    }
    lds_fence();
    return an;
}

__global__ void __launch_bounds__(256, 1)
qmlp_fused_kernel(const float* __restrict__ x,
                  const float* __restrict__ w1, const float* __restrict__ w2,
                  const float* __restrict__ w3, const float* __restrict__ w4,
                  const float* __restrict__ w5,
                  const float* __restrict__ bn0g, const float* __restrict__ bn0b,
                  const float* __restrict__ bn0m, const float* __restrict__ bn0v,
                  const float* __restrict__ bn1g, const float* __restrict__ bn1b,
                  const float* __restrict__ bn1m, const float* __restrict__ bn1v,
                  const float* __restrict__ bn5g, const float* __restrict__ bn5b,
                  const float* __restrict__ bn5m, const float* __restrict__ bn5v,
                  float* __restrict__ out, int B) {
    __shared__ unsigned smem[8 * 256];
    const int lane = threadIdx.x & 31;
    const int l15 = lane & 15;
    const int hi = lane >> 4;
    char* ldsb = (char*)(smem + (threadIdx.x >> 5) * 256);
    float* ldsf = (float*)ldsb;
    const v8i zero = {0, 0, 0, 0, 0, 0, 0, 0};

    v8i B1[4], B2[2], B3[2], B4[2], B5;
    #pragma unroll
    for (int t = 0; t < 4; ++t) B1[t] = build_b(w1, 64, 16, 16 * t, l15, hi);
    #pragma unroll
    for (int t = 0; t < 2; ++t) B2[t] = build_b(w2, 32, 64, 16 * t, l15, hi);
    #pragma unroll
    for (int t = 0; t < 2; ++t) B3[t] = build_b(w3, 32, 32, 16 * t, l15, hi);
    #pragma unroll
    for (int t = 0; t < 2; ++t) B4[t] = build_b(w4, 32, 32, 16 * t, l15, hi);
    B5 = build_b(w5, 5, 32, 0, l15, hi);

    float s0[8], t0[8];
    #pragma unroll
    for (int j = 0; j < 8; ++j) {
        int k = 8 * hi + j;
        float sc = bn0g[k] * rsqrtf(bn0v[k] + BN_EPS);
        s0[j] = sc;
        t0[j] = bn0b[k] - bn0m[k] * sc;
    }
    float s1[4], t1[4];
    #pragma unroll
    for (int j = 0; j < 4; ++j) {
        int n = l15 + 16 * j;
        float sc = bn1g[n] * rsqrtf(bn1v[n] + BN_EPS);
        s1[j] = sc;
        t1[j] = bn1b[n] - bn1m[n] * sc;
    }
    float s5 = 0.0f, t5 = 0.0f;
    if (l15 < 5) {
        float sc = bn5g[l15] * rsqrtf(bn5v[l15] + BN_EPS);
        s5 = sc;
        t5 = bn5b[l15] - bn5m[l15] * sc;
    }

    const int ntiles = (B + 15) >> 4;
    const int nwv = (gridDim.x * blockDim.x) >> 5;
    const int gw = (int)(blockIdx.x * blockDim.x + threadIdx.x) >> 5;

    for (int tile = gw; tile < ntiles; tile += nwv) {
        const int row0 = tile << 4;
        int row = row0 + l15;
        if (row >= B) row = B - 1;

        {
            long nrow = (long)row0 + (long)nwv * 16 + l15;
            if (nrow < (long)B) __builtin_prefetch(x + nrow * 16, 0, 1);
        }

        const float4* xr = (const float4*)(x + (size_t)row * 16 + 8 * hi);
        float4 xa = xr[0], xb = xr[1];
        v8i A = zero;
        A[0] = (int)pk4(q8(fmaxf(xa.x * s0[0] + t0[0], 0.0f)),
                        q8(fmaxf(xa.y * s0[1] + t0[1], 0.0f)),
                        q8(fmaxf(xa.z * s0[2] + t0[2], 0.0f)),
                        q8(fmaxf(xa.w * s0[3] + t0[3], 0.0f)));
        A[1] = (int)pk4(q8(fmaxf(xb.x * s0[4] + t0[4], 0.0f)),
                        q8(fmaxf(xb.y * s0[5] + t0[5], 0.0f)),
                        q8(fmaxf(xb.z * s0[6] + t0[6], 0.0f)),
                        q8(fmaxf(xb.w * s0[7] + t0[7], 0.0f)));

        #pragma unroll
        for (int t = 0; t < 4; ++t) {
            v8i c = wmma_iu8(A, B1[t], zero);
            #pragma unroll
            for (int r = 0; r < 8; ++r) {
                float h = fmaxf((float)c[r] * s1[t] + t1[t], 0.0f);
                ldsb[(r + 8 * hi) * 64 + 16 * t + l15] = (char)q8(h);
            }
        }
        lds_fence();
        v8i A1;
        #pragma unroll
        for (int v = 0; v < 8; ++v) {
            int k0 = 16 * (v >> 1) + 8 * hi + 4 * (v & 1);
            A1[v] = *(const int*)(ldsb + l15 * 64 + k0);
        }
        lds_fence();

        v8i A2 = layer_int(A1, B2, ldsb, l15, hi);
        v8i A3 = layer_int(A2, B3, ldsb, l15, hi);
        v8i A4 = layer_int(A3, B4, ldsb, l15, hi);

        {
            v8i c = wmma_iu8(A4, B5, zero);
            if (l15 < 5) {
                #pragma unroll
                for (int r = 0; r < 8; ++r) {
                    float o = fmaxf((float)c[r] * s5 + t5, 0.0f);
                    ldsf[(r + 8 * hi) * 5 + l15] = o;
                }
            }
        }
        lds_fence();

        float* outp = out + (size_t)row0 * 5;
        int lim = (B - row0) < 16 ? (B - row0) * 5 : 80;
        for (int i = lane; i < lim; i += 32) outp[i] = ldsf[i];
        lds_fence();
    }
}

extern "C" void kernel_launch(void* const* d_in, const int* in_sizes, int n_in,
                              void* d_out, int out_size, void* d_ws, size_t ws_size,
                              hipStream_t stream) {
    const float* x = (const float*)d_in[0];
    const float* w1 = (const float*)d_in[1];
    const float* w2 = (const float*)d_in[2];
    const float* w3 = (const float*)d_in[3];
    const float* w4 = (const float*)d_in[4];
    const float* w5 = (const float*)d_in[5];
    const float* bn0g = (const float*)d_in[6];
    const float* bn0b = (const float*)d_in[7];
    const float* bn0m = (const float*)d_in[8];
    const float* bn0v = (const float*)d_in[9];
    const float* bn1g = (const float*)d_in[10];
    const float* bn1b = (const float*)d_in[11];
    const float* bn1m = (const float*)d_in[12];
    const float* bn1v = (const float*)d_in[13];
    const float* bn5g = (const float*)d_in[14];
    const float* bn5b = (const float*)d_in[15];
    const float* bn5m = (const float*)d_in[16];
    const float* bn5v = (const float*)d_in[17];

    const int B = in_sizes[0] / 16;
    const int ntiles = (B + 15) / 16;
    int blocks = 1024;
    if (blocks * 8 > ntiles) blocks = (ntiles + 7) / 8;
    if (blocks < 1) blocks = 1;

    qmlp_fused_kernel<<<blocks, 256, 0, stream>>>(
        x, w1, w2, w3, w4, w5,
        bn0g, bn0b, bn0m, bn0v,
        bn1g, bn1b, bn1m, bn1v,
        bn5g, bn5b, bn5m, bn5v,
        (float*)d_out, B);
}
